// NormEMAVectorQuantizer_45457933861463
// MI455X (gfx1250) — matrix operand order repaired and checked by reading, not compiled, not run on hardware
//
#include <hip/hip_runtime.h>
#include <hip/hip_bf16.h>

typedef __attribute__((ext_vector_type(16))) _Float16 v16h;
typedef __attribute__((ext_vector_type(8)))  _Float16 v8h;
typedef __attribute__((ext_vector_type(8)))  float    v8f;

#define N_EMBED 8192
#define EMB_DIM 32
#define N_TOKENS 32768
#define DECAY 0.99f

// ---------------- workspace layout (bytes) ----------------
// zn     f32 [B,32]   @ 0            (4 MB)
// zh     f16 [B,32]   @ 4194304      (2 MB)
// wh     f16 [N,32]   @ 6291456      (512 KB)
// wn2    f32 [N]      @ 6815744      (32 KB)
// idx    i32 [B]      @ 6848512      (128 KB)
// bins   f32 [N]      @ 6979584      (32 KB)
// esum   f32 [N,32]   @ 7012352      (1 MB)
// lossacc f32 [1]     @ 8060928
#define WS_ZN    0
#define WS_ZH    4194304
#define WS_WH    6291456
#define WS_WN2   6815744
#define WS_IDX   6848512
#define WS_BINS  6979584
#define WS_ESUM  7012352
#define WS_LOSS  8060928

// ---------- zero the accumulators (bins, esum, lossacc) ----------
__global__ __launch_bounds__(256) void k_init(float* bins, float* esum, float* lossacc) {
    int t = blockIdx.x * blockDim.x + threadIdx.x;
    const int total = N_EMBED * EMB_DIM + N_EMBED + 1;
    if (t >= total) return;
    if (t < N_EMBED * EMB_DIM)      esum[t] = 0.0f;
    else if (t < N_EMBED * EMB_DIM + N_EMBED) bins[t - N_EMBED * EMB_DIM] = 0.0f;
    else                            lossacc[0] = 0.0f;
}

// ---------- normalize z; store f32 + f16 copies (one wave per token) ----------
__global__ __launch_bounds__(256) void k_prep_z(const float* __restrict__ z,
                                                float* __restrict__ zn,
                                                _Float16* __restrict__ zh) {
    int lane = threadIdx.x & 31;
    int tok  = (blockIdx.x * blockDim.x + threadIdx.x) >> 5;
    size_t o = (size_t)tok * EMB_DIM + lane;
    float v = z[o];
    float ss = v * v;
    #pragma unroll
    for (int off = 16; off >= 1; off >>= 1) ss += __shfl_xor(ss, off);
    float nv = v / fmaxf(sqrtf(ss), 1e-12f);
    zn[o] = nv;
    zh[o] = (_Float16)nv;
}

// ---------- f16 weight copy + ||w||^2 per row (one wave per row) ----------
__global__ __launch_bounds__(256) void k_prep_w(const float* __restrict__ weight,
                                                _Float16* __restrict__ wh,
                                                float* __restrict__ wn2) {
    int lane = threadIdx.x & 31;
    int row  = (blockIdx.x * blockDim.x + threadIdx.x) >> 5;
    size_t o = (size_t)row * EMB_DIM + lane;
    float v = weight[o];
    wh[o] = (_Float16)v;
    float ss = v * v;
    #pragma unroll
    for (int off = 16; off >= 1; off >>= 1) ss += __shfl_xor(ss, off);
    if (lane == 0) wn2[row] = ss;
}

// ---------- WMMA GEMM + argmin: one wave owns 16 tokens ----------
// A fragment (16x32 f16, M=token): lanes 0-15 hold K=0..15, lanes 16-31 K=16..31
// B fragment (32x16 f16, N=code):  lanes 0-15 hold K=0..15, lanes 16-31 K=16..31, n = lane%16
// C (16x16 f32): VGPR r, lane L -> M = r + (L>=16 ? 8 : 0), N = L%16
__global__ __launch_bounds__(256) void k_argmin(const _Float16* __restrict__ zh,
                                                const _Float16* __restrict__ wh,
                                                const float* __restrict__ wn2,
                                                int* __restrict__ idxbuf) {
    int lane  = threadIdx.x & 31;
    int wv    = threadIdx.x >> 5;
    int gw    = blockIdx.x * (blockDim.x >> 5) + wv;
    int rowbase = gw * 16;
    int hi    = lane >> 4;      // 0 or 1
    int l15   = lane & 15;

    // load A fragment once; reused for all 512 N-tiles
    const _Float16* zrow = zh + (size_t)(rowbase + l15) * EMB_DIM;
    v8h a0 = *(const v8h*)(zrow + hi * 16);
    v8h a1 = *(const v8h*)(zrow + hi * 16 + 8);
    v16h A = __builtin_shufflevector(a0, a1, 0,1,2,3,4,5,6,7,8,9,10,11,12,13,14,15);

    float best[8];
    int   bidx[8];
    #pragma unroll
    for (int r = 0; r < 8; ++r) { best[r] = 3.4e38f; bidx[r] = 0; }

    for (int nb = 0; nb < N_EMBED; nb += 16) {
        int n = nb + l15;
        // B fragment: lane needs 16 consecutive halves of codebook row n
        v16h Bm = *(const v16h*)(wh + (size_t)n * EMB_DIM + hi * 16);
        v8f C = {};
        C = __builtin_amdgcn_wmma_f32_16x16x32_f16(
                /*neg_a=*/false, A, /*neg_b=*/false, Bm,
                /*c_mod=*/(short)0, C, /*reuse_a=*/false, /*reuse_b=*/false);
        float s = wn2[n];                     // key = ||w||^2 - 2*dot  (minimize)
        #pragma unroll
        for (int r = 0; r < 8; ++r) {
            float key = fmaf(-2.0f, C[r], s);
            bool c = key < best[r];           // per-lane n strictly increases -> first occurrence kept
            best[r] = c ? key : best[r];
            bidx[r] = c ? n   : bidx[r];
        }
    }

    // reduce across the 16 N-lanes in each wave half (xor masks < 16 stay in-half)
    #pragma unroll
    for (int r = 0; r < 8; ++r) {
        float b = best[r];
        int   i = bidx[r];
        #pragma unroll
        for (int off = 1; off < 16; off <<= 1) {
            float ob = __shfl_xor(b, off);
            int   oi = __shfl_xor(i, off);
            bool c = (ob < b) || (ob == b && oi < i);   // argmin: lowest index wins ties
            b = c ? ob : b;
            i = c ? oi : i;
        }
        if (l15 == 0) idxbuf[rowbase + hi * 8 + r] = i;
    }
}

// ---------- gather z_q, scatter EMA stats, loss partial (one wave per token) ----------
__global__ __launch_bounds__(256) void k_scatter(const int* __restrict__ idxbuf,
                                                 const float* __restrict__ weight,
                                                 const float* __restrict__ zn,
                                                 float* __restrict__ o_zq,
                                                 float* __restrict__ o_idx,
                                                 float* __restrict__ bins,
                                                 float* __restrict__ esum,
                                                 float* __restrict__ lossacc) {
    int lane = threadIdx.x & 31;
    int tok  = (blockIdx.x * blockDim.x + threadIdx.x) >> 5;
    int idx  = idxbuf[tok];
    size_t zo = (size_t)tok * EMB_DIM + lane;
    size_t wo = (size_t)idx * EMB_DIM + lane;
    float wv = weight[wo];
    float zv = zn[zo];
    o_zq[zo] = wv;                      // z_q_st == z_q numerically (straight-through)
    float d = wv - zv;
    float sq = d * d;
    #pragma unroll
    for (int off = 16; off >= 1; off >>= 1) sq += __shfl_xor(sq, off);
    atomicAdd(&esum[wo], zv);
    if (lane == 0) {
        atomicAdd(&bins[idx], 1.0f);
        atomicAdd(lossacc, sq);
        o_idx[tok] = (float)idx;
    }
}

// ---------- EMA finalize per codebook row (one wave per row) + loss finalize ----------
__global__ __launch_bounds__(256) void k_finalize(const float* __restrict__ weight,
                                                  const float* __restrict__ cluster_size,
                                                  const float* __restrict__ bins,
                                                  const float* __restrict__ esum,
                                                  const float* __restrict__ lossacc,
                                                  float* __restrict__ o_w,
                                                  float* __restrict__ o_cs,
                                                  float* __restrict__ o_loss) {
    int lane = threadIdx.x & 31;
    int row  = (blockIdx.x * blockDim.x + threadIdx.x) >> 5;
    size_t o = (size_t)row * EMB_DIM + lane;

    float w0 = weight[o];
    float bn = bins[row];
    if (lane == 0) o_cs[row] = cluster_size[row] * DECAY + bn * (1.0f - DECAY);

    float safe = (bn == 0.0f) ? 1.0f : bn;
    float v = esum[o] / safe;
    float ss = v * v;
    #pragma unroll
    for (int off = 16; off >= 1; off >>= 1) ss += __shfl_xor(ss, off);
    float en = v / fmaxf(sqrtf(ss), 1e-12f);
    if (bn == 0.0f) en = w0;

    float nw = w0 * DECAY + en * (1.0f - DECAY);
    float ss2 = nw * nw;
    #pragma unroll
    for (int off = 16; off >= 1; off >>= 1) ss2 += __shfl_xor(ss2, off);
    o_w[o] = nw / fmaxf(sqrtf(ss2), 1e-12f);

    if (blockIdx.x == 0 && threadIdx.x == 0)
        o_loss[0] = lossacc[0] / (float)(N_TOKENS * EMB_DIM);   // BETA = 1
}

extern "C" void kernel_launch(void* const* d_in, const int* in_sizes, int n_in,
                              void* d_out, int out_size, void* d_ws, size_t ws_size,
                              hipStream_t stream) {
    const float* z            = (const float*)d_in[0];
    const float* weight       = (const float*)d_in[1];
    const float* cluster_size = (const float*)d_in[2];

    char* ws = (char*)d_ws;
    float*    zn      = (float*)   (ws + WS_ZN);
    _Float16* zh      = (_Float16*)(ws + WS_ZH);
    _Float16* wh      = (_Float16*)(ws + WS_WH);
    float*    wn2     = (float*)   (ws + WS_WN2);
    int*      idxbuf  = (int*)     (ws + WS_IDX);
    float*    bins    = (float*)   (ws + WS_BINS);
    float*    esum    = (float*)   (ws + WS_ESUM);
    float*    lossacc = (float*)   (ws + WS_LOSS);

    float* out    = (float*)d_out;
    float* o_zq   = out;                                  // [B, 32]
    float* o_loss = o_zq + (size_t)N_TOKENS * EMB_DIM;    // [1]
    float* o_idx  = o_loss + 1;                           // [B]
    float* o_w    = o_idx + N_TOKENS;                     // [N, 32]
    float* o_cs   = o_w + (size_t)N_EMBED * EMB_DIM;      // [N]

    const int zeroTotal = N_EMBED * EMB_DIM + N_EMBED + 1;
    k_init   <<<(zeroTotal + 255) / 256, 256, 0, stream>>>(bins, esum, lossacc);
    k_prep_z <<<(N_TOKENS * 32) / 256,   256, 0, stream>>>(z, zn, zh);
    k_prep_w <<<(N_EMBED * 32) / 256,    256, 0, stream>>>(weight, wh, wn2);
    // one wave per 16 tokens, 8 waves per block -> B/(16*8) blocks
    k_argmin <<<N_TOKENS / 128,          256, 0, stream>>>(zh, wh, wn2, idxbuf);
    k_scatter<<<(N_TOKENS * 32) / 256,   256, 0, stream>>>(idxbuf, weight, zn, o_zq, o_idx,
                                                           bins, esum, lossacc);
    k_finalize<<<(N_EMBED * 32) / 256,   256, 0, stream>>>(weight, cluster_size, bins, esum,
                                                           lossacc, o_w, o_cs, o_loss);
}
